// NTB_13511967113502
// MI455X (gfx1250) — matrix operand order repaired and checked by reading, not compiled, not run on hardware
//
#include <hip/hip_runtime.h>
#include <hip/hip_bf16.h>

// ---------------------------------------------------------------------------
// Transformer block (LN -> QKV -> L2-norm attention -> proj+res -> LN -> MLP)
// for MI455X / gfx1250.  All GEMMs run through v_wmma_f32_16x16x32_bf16
// (bf16 in, f32 accumulate).  Tile staging uses the CDNA5 async
// global->LDS path (global_load_async_to_lds_b128 + s_wait_asynccnt).
// ---------------------------------------------------------------------------

typedef __attribute__((ext_vector_type(16))) __bf16 v16bf;
typedef __attribute__((ext_vector_type(8)))  __bf16 v8bf;
typedef __attribute__((ext_vector_type(8)))  float  v8f;

#define DIM     1024
#define SEQ     2048
#define NTOK    4096      // 2 * 2048
#define HEADS   16
#define HDIM    64
#define MLP_HID 4096

// ---------------- CDNA5 async global->LDS (ASYNCcnt) -----------------------

// Per-lane copy of 16B global -> LDS, tracked by ASYNCcnt.  The LDS operand
// is the low 32 bits of the generic pointer (flat-LDS aperture keeps the
// byte offset in addr[31:0]).
__device__ inline void async_g2l_b128(const __bf16* g, __bf16* l) {
    unsigned int loff = (unsigned int)(unsigned long long)l;
    asm volatile("global_load_async_to_lds_b128 %0, %1, off"
                 :: "v"(loff), "v"((unsigned long long)g) : "memory");
}

__device__ inline void wait_async0() {
#if __has_builtin(__builtin_amdgcn_s_wait_asynccnt)
    __builtin_amdgcn_s_wait_asynccnt(0);
#else
    asm volatile("s_wait_asynccnt 0x0" ::: "memory");
#endif
    __asm__ volatile("" ::: "memory");
}

// ---------------- fragment loaders (match CDNA5 16-bit WMMA VGPR layouts) ---

// A-matrix fragment: lane half 0 holds K = {0..7, 16..23}, half 1 {8..15,
// 24..31}.  Caller passes p already offset by k0 -> chunks at p and p+16.
__device__ inline v16bf frag_split(const __bf16* p) {
    v8bf a = *(const v8bf*)p;
    v8bf b = *(const v8bf*)(p + 16);
    v16bf r;
#pragma unroll
    for (int i = 0; i < 8; ++i) { r[i] = a[i]; r[i + 8] = b[i]; }
    return r;
}

// B-matrix fragment: lane half 0 holds K = {0..7, 16..23}, half 1 {8..15,
// 24..31}.  Caller passes p already offset by k0.
__device__ inline v16bf frag_contig(const __bf16* p) {
    v8bf a = *(const v8bf*)p;
    v8bf b = *(const v8bf*)(p + 16);
    v16bf r;
#pragma unroll
    for (int i = 0; i < 8; ++i) { r[i] = a[i]; r[i + 8] = b[i]; }
    return r;
}

__device__ inline v8f wmma_bf16(v16bf a, v16bf b, v8f c) {
    return __builtin_amdgcn_wmma_f32_16x16x32_bf16(
        false, a, false, b, (short)0, c, false, false);
}

__device__ inline float gelu_exact(float v) {
    return 0.5f * v * (1.0f + erff(v * 0.70710678118654752f));
}

// ---------------- weight / activation conversion ---------------------------

__global__ __launch_bounds__(256) void cvt_f32_bf16(const float* __restrict__ s,
                                                    __bf16* __restrict__ d, int n) {
    for (int i = blockIdx.x * 256 + threadIdx.x; i < n; i += gridDim.x * 256)
        d[i] = (__bf16)s[i];
}

// ---------------- LayerNorm: f32 [tok,1024] -> bf16 ------------------------

__global__ __launch_bounds__(256) void ln_bf16(const float* __restrict__ x,
                                               const float* __restrict__ g,
                                               const float* __restrict__ bta,
                                               __bf16* __restrict__ out) {
    __shared__ float rs[256], rss[256];
    const int tok = blockIdx.x, tid = threadIdx.x;
    const float* xr = x + (size_t)tok * DIM;
    float vals[4], s = 0.f, ss = 0.f;
#pragma unroll
    for (int i = 0; i < 4; ++i) {
        float v = xr[tid + 256 * i];
        vals[i] = v; s += v; ss += v * v;
    }
    rs[tid] = s; rss[tid] = ss;
    __syncthreads();
    for (int off = 128; off > 0; off >>= 1) {
        if (tid < off) { rs[tid] += rs[tid + off]; rss[tid] += rss[tid + off]; }
        __syncthreads();
    }
    const float mu   = rs[0] * (1.0f / DIM);
    const float var  = rss[0] * (1.0f / DIM) - mu * mu;
    const float rinv = rsqrtf(var + 1e-5f);
#pragma unroll
    for (int i = 0; i < 4; ++i) {
        int c = tid + 256 * i;
        out[(size_t)tok * DIM + c] = (__bf16)((vals[i] - mu) * rinv * g[c] + bta[c]);
    }
}

// ---------------- generic WMMA GEMM:  C = A[M,K] * W[N,K]^T ----------------
// Block tile 128(M) x 128(N), K staged 64 at a time, double-buffered LDS
// filled by async global->LDS.  8 waves arranged 4(M) x 2(N); each wave
// owns 32x64 = 2x4 WMMA tiles.  Epilogue behavior is compile-time.

#define BM 128
#define BN 128
#define BK 64
#define LDP 8   // LDS row padding (elements)

template <bool HB, bool HG, bool HR, bool OF, bool OB>
__global__ __launch_bounds__(256) void gemm_bf16(
    const __bf16* __restrict__ A, const __bf16* __restrict__ W,
    float* __restrict__ outf, __bf16* __restrict__ outb,
    const float* __restrict__ bias, const float* __restrict__ resid,
    int M, int N, int K) {
    __shared__ __align__(16) __bf16 Al[2][BM][BK + LDP];
    __shared__ __align__(16) __bf16 Wl[2][BN][BK + LDP];

    const int tid  = threadIdx.x;
    const int lane = tid & 31, w = tid >> 5;
    const int tM = blockIdx.y * BM, tN = blockIdx.x * BN;
    const int wm = (w & 3) * 32, wn = (w >> 2) * 64;

    const int srow = tid >> 3;          // staging row, 32 rows apart x4
    const int scol = (tid & 7) * 8;     // staging col chunk (8 bf16 = 16B)
    const __bf16* Ag = A + (size_t)(tM + srow) * K + scol;
    const __bf16* Wg = W + (size_t)(tN + srow) * K + scol;

    const v8f z8 = {0.f, 0.f, 0.f, 0.f, 0.f, 0.f, 0.f, 0.f};
    v8f acc[2][4];
#pragma unroll
    for (int mi = 0; mi < 2; ++mi)
#pragma unroll
        for (int ni = 0; ni < 4; ++ni) acc[mi][ni] = z8;

    // prologue: async-stage k-step 0 into buffer 0
#pragma unroll
    for (int i = 0; i < 4; ++i) {
        async_g2l_b128(&Ag[(size_t)(32 * i) * K], &Al[0][srow + 32 * i][scol]);
        async_g2l_b128(&Wg[(size_t)(32 * i) * K], &Wl[0][srow + 32 * i][scol]);
    }
    wait_async0();
    __syncthreads();

    const int nk = K / BK;
    for (int ks = 0; ks < nk; ++ks) {
        const int buf = ks & 1;
        // async-stage next k-step into the free buffer while computing
        if (ks + 1 < nk) {
            const size_t koff = (size_t)(ks + 1) * BK;
#pragma unroll
            for (int i = 0; i < 4; ++i) {
                async_g2l_b128(&Ag[(size_t)(32 * i) * K + koff],
                               &Al[1 - buf][srow + 32 * i][scol]);
                async_g2l_b128(&Wg[(size_t)(32 * i) * K + koff],
                               &Wl[1 - buf][srow + 32 * i][scol]);
            }
        }
#pragma unroll
        for (int kk = 0; kk < BK; kk += 32) {
            const int ak0 = kk + ((lane < 16) ? 0 : 8);
            const int bk0 = kk + ((lane < 16) ? 0 : 8);
            v16bf afr[2], bfr[4];
#pragma unroll
            for (int mi = 0; mi < 2; ++mi)
                afr[mi] = frag_split(&Al[buf][wm + mi * 16 + (lane & 15)][ak0]);
#pragma unroll
            for (int ni = 0; ni < 4; ++ni)
                bfr[ni] = frag_contig(&Wl[buf][wn + ni * 16 + (lane & 15)][bk0]);
#pragma unroll
            for (int mi = 0; mi < 2; ++mi)
#pragma unroll
                for (int ni = 0; ni < 4; ++ni)
                    acc[mi][ni] = wmma_bf16(afr[mi], bfr[ni], acc[mi][ni]);
        }
        if (ks + 1 < nk) wait_async0();
        __syncthreads();
    }

    // epilogue: C layout -> VGPR r holds (M = r + 8*(lane/16), N = lane&15)
    const int rowsel = (lane >> 4) * 8;
    const int ncol   = lane & 15;
#pragma unroll
    for (int ni = 0; ni < 4; ++ni) {
        const int col = tN + wn + ni * 16 + ncol;
        float bv = 0.f;
        if (HB) bv = bias[col];
#pragma unroll
        for (int mi = 0; mi < 2; ++mi)
#pragma unroll
            for (int r = 0; r < 8; ++r) {
                const int row = tM + wm + mi * 16 + r + rowsel;
                float v = acc[mi][ni][r];
                if (HB) v += bv;
                if (HG) v = gelu_exact(v);
                if (HR) v += resid[(size_t)row * N + col];
                if (OF) outf[(size_t)row * N + col] = v;
                if (OB) outb[(size_t)row * N + col] = (__bf16)v;
            }
    }
}

// ---------------- q/k L2-normalize + repack to [bh][n][d] bf16 -------------
// One wave per (token, head): 32 lanes x 2 elements = 64-dim head.

__global__ __launch_bounds__(256) void qkv_norm(const float* __restrict__ qkv,
                                                __bf16* __restrict__ qn,
                                                __bf16* __restrict__ kn,
                                                __bf16* __restrict__ vn) {
    const int gw   = blockIdx.x * 8 + (threadIdx.x >> 5);
    const int lane = threadIdx.x & 31;
    const int tok = gw >> 4, h = gw & 15;
    const int b = tok >> 11, n = tok & 2047;
    const float* p = qkv + (size_t)tok * (3 * DIM) + h * HDIM;
    const int d0 = lane * 2;

    float q0 = p[d0],           q1 = p[d0 + 1];
    float k0 = p[DIM + d0],     k1 = p[DIM + d0 + 1];
    float v0 = p[2 * DIM + d0], v1 = p[2 * DIM + d0 + 1];
    float sq = q0 * q0 + q1 * q1;
    float sk = k0 * k0 + k1 * k1;
#pragma unroll
    for (int m = 1; m < 32; m <<= 1) {
        sq += __shfl_xor(sq, m, 32);
        sk += __shfl_xor(sk, m, 32);
    }
    const float qden = 1.0f / fmaxf(sqrtf(sq), 1e-12f);
    const float kden = 1.0f / fmaxf(sqrtf(sk), 1e-12f);
    const size_t o = (((size_t)(b * HEADS + h)) * SEQ + n) * HDIM + d0;
    qn[o] = (__bf16)(q0 * qden); qn[o + 1] = (__bf16)(q1 * qden);
    kn[o] = (__bf16)(k0 * kden); kn[o + 1] = (__bf16)(k1 * kden);
    vn[o] = (__bf16)v0;          vn[o + 1] = (__bf16)v1;
}

// ---------------- flash attention, 64-key steps, double-buffered -----------
// grid = (16 q-chunks, 32 bh).  8 waves/block, wave owns 16 q rows.
// K tiles staged via async global->LDS; V staged through registers (needs
// the transpose scatter).  One shuffle reduction per row per 64 keys.

#define KT 64   // keys per iteration

__global__ __launch_bounds__(256) void attn_fwd(const __bf16* __restrict__ Q,
                                                const __bf16* __restrict__ Kb,
                                                const __bf16* __restrict__ V,
                                                const float* __restrict__ temp,
                                                __bf16* __restrict__ out) {
    __shared__ __align__(16) __bf16 Kl[2][KT][HDIM + LDP];   // keys x d
    __shared__ __align__(16) __bf16 Vt[2][HDIM][KT + LDP];   // d x keys
    __shared__ __align__(16) __bf16 Pl[8][16][KT + LDP];     // per-wave P tile

    const int tid = threadIdx.x, lane = tid & 31, w = tid >> 5;
    const int bh = blockIdx.y, b = bh >> 4, h = bh & 15;
    const int qb = blockIdx.x * 128 + w * 16;
    const size_t base = (size_t)bh * SEQ * HDIM;
    const float tmp = temp[h];

    const int mrow16 = lane & 15;
    const int ak0 = (lane < 16) ? 0 : 8;    // A-fragment K offset
    const int bk0 = (lane < 16) ? 0 : 8;    // B-fragment K offset

    // staging coordinates: 64 rows x 64 cols = 512 16B chunks, 2 per thread
    const int sr0 = tid >> 3, sr1 = (tid + 256) >> 3;   // rows 0..63
    const int sc  = (tid & 7) * 8;                      // col chunk

    // Q fragments (row-major [n][d] -> contiguous chunks along d)
    v16bf qf[2];
    qf[0] = frag_split(&Q[base + (size_t)(qb + mrow16) * HDIM + ak0]);
    qf[1] = frag_split(&Q[base + (size_t)(qb + mrow16) * HDIM + 32 + ak0]);

    const v8f z8 = {0.f, 0.f, 0.f, 0.f, 0.f, 0.f, 0.f, 0.f};
    v8f o[4];
    float mrow[8], lrow[8];
#pragma unroll
    for (int t = 0; t < 4; ++t) o[t] = z8;
#pragma unroll
    for (int r = 0; r < 8; ++r) { mrow[r] = -1e30f; lrow[r] = 0.f; }

    v8bf vr[2];
    // prologue: async-stage K tile 0; V tile 0 via registers (transposed)
    async_g2l_b128(&Kb[base + (size_t)sr0 * HDIM + sc], &Kl[0][sr0][sc]);
    async_g2l_b128(&Kb[base + (size_t)sr1 * HDIM + sc], &Kl[0][sr1][sc]);
    vr[0] = *(const v8bf*)&V[base + (size_t)sr0 * HDIM + sc];
    vr[1] = *(const v8bf*)&V[base + (size_t)sr1 * HDIM + sc];
#pragma unroll
    for (int e = 0; e < 8; ++e) {
        Vt[0][sc + e][sr0] = vr[0][e];
        Vt[0][sc + e][sr1] = vr[1][e];
    }
    wait_async0();
    __syncthreads();

    const int nkt = SEQ / KT;
    for (int it = 0; it < nkt; ++it) {
        const int buf = it & 1;
        // prefetch next key tile while computing this one
        if (it + 1 < nkt) {
            const size_t koff = base + (size_t)(it + 1) * KT * HDIM;
            async_g2l_b128(&Kb[koff + (size_t)sr0 * HDIM + sc], &Kl[1 - buf][sr0][sc]);
            async_g2l_b128(&Kb[koff + (size_t)sr1 * HDIM + sc], &Kl[1 - buf][sr1][sc]);
            vr[0] = *(const v8bf*)&V[koff + (size_t)sr0 * HDIM + sc];
            vr[1] = *(const v8bf*)&V[koff + (size_t)sr1 * HDIM + sc];
        }

        // S = Q K^T : four 16x16 tiles covering 64 keys
        v8f s[4];
#pragma unroll
        for (int ni = 0; ni < 4; ++ni) {
            s[ni] = z8;
#pragma unroll
            for (int ks = 0; ks < 2; ++ks) {
                v16bf kf = frag_contig(&Kl[buf][ni * 16 + mrow16][ks * 32 + bk0]);
                s[ni] = wmma_bf16(qf[ks], kf, s[ni]);
            }
        }

        // online softmax: one shuffle reduction per row per 64 keys
#pragma unroll
        for (int r = 0; r < 8; ++r) {
            float sv[4];
#pragma unroll
            for (int ni = 0; ni < 4; ++ni) sv[ni] = s[ni][r] * tmp;
            float rm = fmaxf(fmaxf(sv[0], sv[1]), fmaxf(sv[2], sv[3]));
#pragma unroll
            for (int m = 1; m < 16; m <<= 1) rm = fmaxf(rm, __shfl_xor(rm, m, 32));
            float mn = fmaxf(mrow[r], rm);
            float al = __expf(mrow[r] - mn);
            float pv[4], rs = 0.f;
#pragma unroll
            for (int ni = 0; ni < 4; ++ni) { pv[ni] = __expf(sv[ni] - mn); rs += pv[ni]; }
#pragma unroll
            for (int m = 1; m < 16; m <<= 1) rs += __shfl_xor(rs, m, 32);
            lrow[r] = lrow[r] * al + rs;
            mrow[r] = mn;
#pragma unroll
            for (int t = 0; t < 4; ++t) o[t][r] *= al;
            const int M_ = r + ((lane >> 4) * 8);
#pragma unroll
            for (int ni = 0; ni < 4; ++ni)
                Pl[w][M_][ni * 16 + mrow16] = (__bf16)pv[ni];
        }
        __asm__ volatile("" ::: "memory");
        __builtin_amdgcn_wave_barrier();   // wave-private LDS; LDS is in-order

        // O += P V : P is 16x64 A-matrix (2 k-steps), Vt rows are contiguous
        v16bf pf[2];
        pf[0] = frag_split(&Pl[w][mrow16][ak0]);
        pf[1] = frag_split(&Pl[w][mrow16][32 + ak0]);
#pragma unroll
        for (int t = 0; t < 4; ++t)
#pragma unroll
            for (int ps = 0; ps < 2; ++ps) {
                v16bf vf = frag_contig(&Vt[buf][t * 16 + mrow16][ps * 32 + bk0]);
                o[t] = wmma_bf16(pf[ps], vf, o[t]);
            }

        // stage the prefetched V tile; wait for the async K tile
        if (it + 1 < nkt) {
#pragma unroll
            for (int e = 0; e < 8; ++e) {
                Vt[1 - buf][sc + e][sr0] = vr[0][e];
                Vt[1 - buf][sc + e][sr1] = vr[1][e];
            }
            wait_async0();
        }
        __syncthreads();
    }

    // normalize & write [tok, h*64 + d] bf16 (feeds proj GEMM as A)
#pragma unroll
    for (int r = 0; r < 8; ++r) {
        int row = qb + r + ((lane >> 4) * 8);
        float inv = 1.0f / lrow[r];
        size_t obase = ((size_t)(b * SEQ + row)) * DIM + h * HDIM;
#pragma unroll
        for (int t = 0; t < 4; ++t)
            out[obase + t * 16 + mrow16] = (__bf16)(o[t][r] * inv);
    }
}

// ---------------------------------------------------------------------------

extern "C" void kernel_launch(void* const* d_in, const int* in_sizes, int n_in,
                              void* d_out, int out_size, void* d_ws, size_t ws_size,
                              hipStream_t stream) {
    const float* x      = (const float*)d_in[0];
    const float* ln1_g  = (const float*)d_in[1];
    const float* ln1_b  = (const float*)d_in[2];
    const float* w_qkv  = (const float*)d_in[3];
    const float* w_proj = (const float*)d_in[4];
    const float* temp   = (const float*)d_in[5];
    const float* ln2_g  = (const float*)d_in[6];
    const float* ln2_b  = (const float*)d_in[7];
    const float* w1     = (const float*)d_in[8];
    const float* b1     = (const float*)d_in[9];
    const float* w2     = (const float*)d_in[10];
    const float* b2     = (const float*)d_in[11];
    float* out = (float*)d_out;

    char* ws = (char*)d_ws;
    __bf16* wqkv_b  = (__bf16*)(ws + 0);                      // 3072*1024 bf16
    __bf16* wproj_b = (__bf16*)(ws + 6291456);                // 1024*1024
    __bf16* w1_b    = (__bf16*)(ws + 8388608);                // 4096*1024
    __bf16* w2_b    = (__bf16*)(ws + 16777216);               // 1024*4096
    __bf16* h1      = (__bf16*)(ws + 25165824);               // LN1 out  [4096,1024]
    float*  qkvf    = (float*) (ws + 33554432);               // [4096,3072] f32
    __bf16* qn      = (__bf16*)(ws + 83886080);               // [32,2048,64]
    __bf16* kn      = (__bf16*)(ws + 92274688);
    __bf16* vn      = (__bf16*)(ws + 100663296);
    __bf16* attnb   = (__bf16*)(ws + 109051904);              // [4096,1024]
    float*  x1      = (float*) (ws + 117440512);              // residual stream f32
    __bf16* h2      = (__bf16*)(ws + 134217728);              // LN2 out
    __bf16* hid     = (__bf16*)(ws + 142606336);              // GELU out [4096,4096]

    // 1) weights -> bf16
    cvt_f32_bf16<<<2048, 256, 0, stream>>>(w_qkv,  wqkv_b,  3 * DIM * DIM);
    cvt_f32_bf16<<<1024, 256, 0, stream>>>(w_proj, wproj_b, DIM * DIM);
    cvt_f32_bf16<<<2048, 256, 0, stream>>>(w1,     w1_b,    MLP_HID * DIM);
    cvt_f32_bf16<<<2048, 256, 0, stream>>>(w2,     w2_b,    DIM * MLP_HID);

    // 2) LN1
    ln_bf16<<<NTOK, 256, 0, stream>>>(x, ln1_g, ln1_b, h1);

    // 3) QKV GEMM: [4096,1024] x [3072,1024]^T -> f32
    gemm_bf16<false, false, false, true, false>
        <<<dim3(3 * DIM / BN, NTOK / BM), 256, 0, stream>>>(
            h1, wqkv_b, qkvf, nullptr, nullptr, nullptr, NTOK, 3 * DIM, DIM);

    // 4) L2-normalize q,k; repack q,k,v
    qkv_norm<<<NTOK * HEADS / 8, 256, 0, stream>>>(qkvf, qn, kn, vn);

    // 5) attention
    attn_fwd<<<dim3(SEQ / 128, 2 * HEADS), 256, 0, stream>>>(qn, kn, vn, temp, attnb);

    // 6) proj + residual(x) -> x1 (f32)
    gemm_bf16<false, false, true, true, false>
        <<<dim3(DIM / BN, NTOK / BM), 256, 0, stream>>>(
            attnb, wproj_b, x1, nullptr, nullptr, x, NTOK, DIM, DIM);

    // 7) LN2
    ln_bf16<<<NTOK, 256, 0, stream>>>(x1, ln2_g, ln2_b, h2);

    // 8) MLP up + bias + GELU -> bf16 hid
    gemm_bf16<true, true, false, false, true>
        <<<dim3(MLP_HID / BN, NTOK / BM), 256, 0, stream>>>(
            h2, w1_b, nullptr, hid, b1, nullptr, NTOK, MLP_HID, DIM);

    // 9) MLP down + bias + residual(x1) -> d_out (f32)
    gemm_bf16<true, false, true, true, false>
        <<<dim3(DIM / BN, NTOK / BM), 256, 0, stream>>>(
            hid, w2_b, out, nullptr, b2, x1, NTOK, DIM, MLP_HID);
}
